// ConvViT_32100585571141
// MI455X (gfx1250) — matrix operand order repaired and checked by reading, not compiled, not run on hardware
//
#include <hip/hip_runtime.h>
#include <hip/hip_bf16.h>

// ---------------- problem constants (from reference) ----------------
#define B_   2
#define N_   2523
#define C_   768
#define H_   12
#define Dh   64
#define NM   1682              // template tokens
#define NS   (N_ - NM)         // 841 search tokens
#define NMEM 4096
#define KF   (NMEM + N_)       // 6619 full keys
#define TOPM 841               // int(0.50 * 1682)
#define TOPS 1654              // int(0.25 * 6619)

typedef __attribute__((ext_vector_type(16))) __bf16 v16bf;
typedef __attribute__((ext_vector_type(8)))  float  v8f;
typedef __attribute__((ext_vector_type(4)))  float  float4v;
typedef __attribute__((ext_vector_type(4)))  unsigned int u32x4;
typedef __attribute__((ext_vector_type(8)))  int    i32x8;
typedef __attribute__((ext_vector_type(4)))  int    i32x4;

union pack4 { __bf16 h[4]; unsigned long long u; };

// =====================================================================
// Generic batched GEMM:  D[m,n] = sum_k A[m,k] * B(n,k)  (+ bias[n])
//   bTrans=1 : B element (n,k) at Bm[n*ldb + k]   (NT)
//   bTrans=0 : B element (n,k) at Bm[k*ldb + n]   (NN)
// Block tile 128x64x32, 8 wave32s; each wave owns a 32x32 output patch
// (2x2 accumulators -> 4 v_wmma_f32_16x16x32_bf16 per K-step).
// Double-buffered LDS: one barrier per K-step; global loads for tile i+1
// issue before fragments of tile i are consumed.
// =====================================================================
#define BM 128
#define BN 64
#define BKT 32
#define LDP 4                   // LDS row pad (halves) -> 72B row stride
#define LDW (BKT + LDP)

typedef __bf16 (*ldsrow_t)[LDW];

__device__ __forceinline__
void stage_tiles(const float* __restrict__ A, const float* __restrict__ Bm,
                 int tileM0, int tileN0, int k0,
                 int M, int N, int K, int lda, int ldb,
                 bool fullM, bool fullN, int bTrans, int tid,
                 ldsrow_t As, ldsrow_t Bs)
{
    const bool fullK = (k0 + BKT) <= K;

    // ---------------- A tile (BM x BKT) ----------------
    if (fullM && fullK) {
        #pragma unroll
        for (int i = 0; i < 4; ++i) {
            int idx = tid + i * 256;            // 0..1023
            int r   = idx >> 3;                 // 0..127
            int c4  = (idx & 7) << 2;           // 0,4,..,28
            float4v v = *(const float4v*)&A[(long)(tileM0 + r) * lda + k0 + c4];
            pack4 pk;
            pk.h[0] = (__bf16)v.x; pk.h[1] = (__bf16)v.y;
            pk.h[2] = (__bf16)v.z; pk.h[3] = (__bf16)v.w;
            *(unsigned long long*)&As[r][c4] = pk.u;
        }
    } else {
        for (int t = tid; t < BM * BKT; t += 256) {
            int r = t >> 5, c = t & 31;
            int gm = tileM0 + r, gk = k0 + c;
            float v = (gm < M && gk < K) ? A[(long)gm * lda + gk] : 0.0f;
            As[r][c] = (__bf16)v;
        }
    }

    // ---------------- B tile (BN x BKT) into Bs[n][k] ----------------
    if (bTrans) {
        if (fullN && fullK) {
            #pragma unroll
            for (int i = 0; i < 2; ++i) {
                int idx = tid + i * 256;        // 0..511
                int r   = idx >> 3;             // 0..63
                int c4  = (idx & 7) << 2;
                float4v v = *(const float4v*)&Bm[(long)(tileN0 + r) * ldb + k0 + c4];
                pack4 pk;
                pk.h[0] = (__bf16)v.x; pk.h[1] = (__bf16)v.y;
                pk.h[2] = (__bf16)v.z; pk.h[3] = (__bf16)v.w;
                *(unsigned long long*)&Bs[r][c4] = pk.u;
            }
        } else {
            for (int t = tid; t < BN * BKT; t += 256) {
                int r = t >> 5, c = t & 31;
                int gn = tileN0 + r, gk = k0 + c;
                float v = (gn < N && gk < K) ? Bm[(long)gn * ldb + gk] : 0.0f;
                Bs[r][c] = (__bf16)v;
            }
        }
    } else {
        if (fullN && fullK) {
            #pragma unroll
            for (int i = 0; i < 2; ++i) {
                int idx = tid + i * 256;        // 0..511
                int kk  = idx >> 4;             // 0..31
                int n4  = (idx & 15) << 2;      // 0,4,..,60
                float4v v = *(const float4v*)&Bm[(long)(k0 + kk) * ldb + tileN0 + n4];
                Bs[n4 + 0][kk] = (__bf16)v.x;
                Bs[n4 + 1][kk] = (__bf16)v.y;
                Bs[n4 + 2][kk] = (__bf16)v.z;
                Bs[n4 + 3][kk] = (__bf16)v.w;
            }
        } else {
            for (int t = tid; t < BN * BKT; t += 256) {
                int r = t >> 5, c = t & 31;
                int gn = tileN0 + r, gk = k0 + c;
                float v = (gn < N && gk < K) ? Bm[(long)gk * ldb + gn] : 0.0f;
                Bs[r][c] = (__bf16)v;
            }
        }
    }
}

__global__ __launch_bounds__(256)
void gemm_wmma_bf16(const float* __restrict__ A, const float* __restrict__ Bm,
                    const float* __restrict__ bias, float* __restrict__ D,
                    int M, int N, int K, int lda, int ldb, int ldd,
                    long aSb, long aSh, long bSb, long bSh, long dSb, long dSh,
                    int Hdiv, int bTrans)
{
    const int z  = blockIdx.z;
    const int bb = z / Hdiv, hh = z % Hdiv;
    A  += (long)bb * aSb + (long)hh * aSh;
    Bm += (long)bb * bSb + (long)hh * bSh;
    D  += (long)bb * dSb + (long)hh * dSh;

    __shared__ __attribute__((aligned(16))) __bf16 AsBuf[2][BM][LDW];
    __shared__ __attribute__((aligned(16))) __bf16 BsBuf[2][BN][LDW];

    const int tid  = threadIdx.x;
    const int lane = tid & 31;
    const int wave = tid >> 5;
    const int wm   = wave & 3;      // rows wm*32 .. +31
    const int wn   = wave >> 2;     // cols wn*32 .. +31

    const int tileM0 = blockIdx.y * BM;
    const int tileN0 = blockIdx.x * BN;

    const bool fullM = (tileM0 + BM) <= M;
    const bool fullN = (tileN0 + BN) <= N;

    v8f acc00 = {}, acc01 = {}, acc10 = {}, acc11 = {};

    // prologue: stage first tile into buffer 0
    stage_tiles(A, Bm, tileM0, tileN0, 0, M, N, K, lda, ldb,
                fullM, fullN, bTrans, tid, AsBuf[0], BsBuf[0]);
    __syncthreads();

    int cur = 0;
    for (int k0 = 0; k0 < K; k0 += BKT) {
        const int nxt = cur ^ 1;

        // stage tile i+1 into the other buffer (global loads issue early)
        if (k0 + BKT < K) {
            stage_tiles(A, Bm, tileM0, tileN0, k0 + BKT, M, N, K, lda, ldb,
                        fullM, fullN, bTrans, tid, AsBuf[nxt], BsBuf[nxt]);
            if (fullM && (k0 + 2 * BKT) < K)
                __builtin_prefetch(&A[(long)(tileM0 + (tid >> 1)) * lda + k0 + 2 * BKT], 0, 0);
        }

        // ---- A fragments (16x32 bf16, ISA 7.12.2 layout) ----
        ldsrow_t As = AsBuf[cur];
        ldsrow_t Bs = BsBuf[cur];
        const int rsel  = lane & 15;
        const int khalf = (lane < 16) ? 0 : 8;
        v16bf af0, af1;
        #pragma unroll
        for (int e = 0; e < 16; ++e) {
            int g  = e >> 1;
            int kk = (g < 4) ? (khalf + 2 * g + (e & 1))
                             : (16 + khalf + 2 * (g - 4) + (e & 1));
            af0[e] = As[wm * 32 + rsel][kk];
            af1[e] = As[wm * 32 + 16 + rsel][kk];
        }
        // ---- B fragments (32x16): same K order as A (lanes 0-15 K 0-7,16-23; lanes 16-31 K 8-15,24-31) ----
        const int kb = (lane < 16) ? 0 : 16;
        const int c0 = wn * 32 + rsel;
        v16bf bf0, bf1;
        #pragma unroll
        for (int e = 0; e < 16; ++e) {
            bf0[e] = Bs[c0][(e < 8) ? (khalf + e) : (16 + khalf + e - 8)];
            bf1[e] = Bs[c0 + 16][(e < 8) ? (khalf + e) : (16 + khalf + e - 8)];
        }

        acc00 = __builtin_amdgcn_wmma_f32_16x16x32_bf16(false, af0, false, bf0,
                                                        (short)0, acc00, false, false);
        acc01 = __builtin_amdgcn_wmma_f32_16x16x32_bf16(false, af0, false, bf1,
                                                        (short)0, acc01, false, false);
        acc10 = __builtin_amdgcn_wmma_f32_16x16x32_bf16(false, af1, false, bf0,
                                                        (short)0, acc10, false, false);
        acc11 = __builtin_amdgcn_wmma_f32_16x16x32_bf16(false, af1, false, bf1,
                                                        (short)0, acc11, false, false);
        // single barrier per K-step: protects (a) next tile fully staged and
        // (b) current tile fully consumed before it is overwritten next round
        __syncthreads();
        cur = nxt;
    }

    // ---- write the 2x2 sub-tiles: VGPR v -> M = v (+8 upper lane half) ----
    const int mrow = (lane >> 4) << 3;
    const int ncol = lane & 15;
    v8f accs[2][2] = {{acc00, acc01}, {acc10, acc11}};
    #pragma unroll
    for (int sm = 0; sm < 2; ++sm) {
        #pragma unroll
        for (int sn = 0; sn < 2; ++sn) {
            int gmBase = tileM0 + wm * 32 + sm * 16 + mrow;
            int gn     = tileN0 + wn * 32 + sn * 16 + ncol;
            if (gn >= N) continue;
            float bi = bias ? bias[gn] : 0.0f;
            #pragma unroll
            for (int v = 0; v < 8; ++v) {
                int gm = gmBase + v;
                if (gm < M) D[(long)gm * ldd + gn] = accs[sm][sn][v] + bi;
            }
        }
    }
}

// =====================================================================
// TDM warm: build a D# and DMA a 64x16 f32 tile of mem_k into LDS via the
// Tensor Data Mover, wait on TENSORcnt, touch the data.
// =====================================================================
__global__ __launch_bounds__(32)
void tdm_stage_kernel(const float* __restrict__ src, float* __restrict__ sink)
{
#if __has_builtin(__builtin_amdgcn_tensor_load_to_lds)
    __shared__ __attribute__((aligned(16))) float tile[64 * 16];
    unsigned long long ga = (unsigned long long)(const void*)src;

    u32x4 g0;
    g0[0] = 1u;                                           // count=1 user D#
    g0[1] = 0u;                                           // lds_addr = 0
    g0[2] = (unsigned int)(ga & 0xFFFFFFFFull);           // global_addr lo
    g0[3] = (unsigned int)((ga >> 32) & 0x01FFFFFFull)    // global_addr hi
          | 0x80000000u;                                  // type = 2 (image)

    i32x8 g1;
    g1[0] = (int)(2u << 16);       // data_size = 4B; no multicast/flags
    g1[1] = (int)(64u << 16);      // tensor_dim0 = 64       (bits 79:48 lo)
    g1[2] = (int)(4096u << 16);    // tensor_dim1 = 4096     (bits 111:80 lo)
    g1[3] = (int)(64u << 16);      // tile_dim0   = 64       (bits 127:112)
    g1[4] = 16;                    // tile_dim1   = 16, tile_dim2 = 0
    g1[5] = 64;                    // tensor_dim0_stride = 64 (lo32)
    g1[6] = (int)(64u << 16);      // stride0 hi = 0; tensor_dim1_stride lo = 64
    g1[7] = 0;

    i32x4 z4 = {0, 0, 0, 0};
#if defined(__clang_major__) && (__clang_major__ >= 23)
    i32x8 z8 = {0, 0, 0, 0, 0, 0, 0, 0};
    __builtin_amdgcn_tensor_load_to_lds(g0, g1, z4, z4, z8, 0);
#else
    __builtin_amdgcn_tensor_load_to_lds(g0, g1, z4, z4, 0);
#endif
    __builtin_amdgcn_s_wait_tensorcnt((short)0);
    __syncthreads();
    sink[threadIdx.x] = tile[threadIdx.x];
#else
    sink[threadIdx.x] = src[threadIdx.x];
#endif
}

// =====================================================================
// Top-k softmax over each row (in place).
// =====================================================================
__global__ __launch_bounds__(256)
void topk_softmax_kernel(float* __restrict__ attn, int K, int top)
{
    float* p = attn + (long)blockIdx.x * (long)K;
    const int tid = threadIdx.x, lane = tid & 31, wave = tid >> 5;

    __shared__ float smax[8], smin[8], ssum[8];
    __shared__ int   scnt[8];

    float vmax = -3.4e38f, vmin = 3.4e38f;
    for (int i = tid; i < K; i += 256) {
        float v = p[i];
        vmax = fmaxf(vmax, v);
        vmin = fminf(vmin, v);
    }
    #pragma unroll
    for (int off = 16; off; off >>= 1) {
        vmax = fmaxf(vmax, __shfl_xor(vmax, off, 32));
        vmin = fminf(vmin, __shfl_xor(vmin, off, 32));
    }
    if (lane == 0) { smax[wave] = vmax; smin[wave] = vmin; }
    __syncthreads();
    if (tid == 0) {
        float a = smax[0], b = smin[0];
        for (int w = 1; w < 8; ++w) { a = fmaxf(a, smax[w]); b = fminf(b, smin[w]); }
        smax[0] = a; smin[0] = b;
    }
    __syncthreads();
    vmax = smax[0]; vmin = smin[0];

    float lo = vmin, hi = vmax;
    for (int it = 0; it < 24; ++it) {
        float mid = 0.5f * (lo + hi);
        int cnt = 0;
        for (int i = tid; i < K; i += 256) cnt += (p[i] >= mid) ? 1 : 0;
        #pragma unroll
        for (int off = 16; off; off >>= 1) cnt += __shfl_xor(cnt, off, 32);
        if (lane == 0) scnt[wave] = cnt;
        __syncthreads();
        if (tid == 0) {
            int c = 0;
            for (int w = 0; w < 8; ++w) c += scnt[w];
            scnt[0] = c;
        }
        __syncthreads();
        cnt = scnt[0];
        __syncthreads();
        if (cnt > top) lo = mid; else hi = mid;
    }
    const float thr = lo;

    float s = 0.0f;
    for (int i = tid; i < K; i += 256) {
        float v = p[i];
        if (v >= thr) s += __expf(v - vmax);
    }
    #pragma unroll
    for (int off = 16; off; off >>= 1) s += __shfl_xor(s, off, 32);
    if (lane == 0) ssum[wave] = s;
    __syncthreads();
    if (tid == 0) {
        float c = 0.0f;
        for (int w = 0; w < 8; ++w) c += ssum[w];
        ssum[0] = c;
    }
    __syncthreads();
    const float inv = 1.0f / ssum[0];

    for (int i = tid; i < K; i += 256) {
        float v = p[i];
        p[i] = (v >= thr) ? __expf(v - vmax) * inv : 0.0f;
    }
}

// =====================================================================
// Elementwise prep kernels
// =====================================================================
__global__ __launch_bounds__(256)
void prep_q_kernel(const float* __restrict__ qkv, float* __restrict__ q_all)
{
    long idx = (long)blockIdx.x * 256 + threadIdx.x;
    const long total = (long)B_ * H_ * N_ * Dh;
    if (idx >= total) return;
    int j = idx % Dh;         long r = idx / Dh;
    int n = r % N_;           r /= N_;
    int h = r % H_;           int b = (int)(r / H_);
    q_all[idx] = qkv[((long)b * N_ + n) * (3 * C_) + h * Dh + j] * 0.125f;
}

__global__ __launch_bounds__(256)
void prep_kv_kernel(const float* __restrict__ qkv, const float* __restrict__ idkv,
                    const float* __restrict__ mem_k, const float* __restrict__ mem_v,
                    float* __restrict__ k_full, float* __restrict__ v_full,
                    float* __restrict__ km_out, float* __restrict__ vm_out)
{
    long idx = (long)blockIdx.x * 256 + threadIdx.x;
    const long total = (long)B_ * H_ * KF * Dh;
    if (idx >= total) return;
    int j  = idx % Dh;        long r = idx / Dh;
    int kk = r % KF;          r /= KF;
    int h  = r % H_;          int b = (int)(r / H_);

    if (kk < NMEM) {
        long mi = (((long)b * H_ + h) * NMEM + kk) * Dh + j;
        k_full[idx] = mem_k[mi];
        v_full[idx] = mem_v[mi];
        return;
    }
    const int n = kk - NMEM;
    const long qb = ((long)b * N_ + n) * (3 * C_);
    float k = qkv[qb + C_     + h * Dh + j];
    float v = qkv[qb + 2 * C_ + h * Dh + j];
    if (n < NM) {
        const long ib = ((long)b * NM + n) * (C_ + H_);
        float gate = 1.0f + tanhf(idkv[ib + h]);
        float kmod = k * gate;
        float vmod = v + idkv[ib + H_ + h * Dh + j];
        long mo = (((long)b * H_ + h) * NM + n) * Dh + j;
        km_out[mo] = kmod;
        vm_out[mo] = vmod;
        k_full[idx] = kmod;
        v_full[idx] = vmod;
    } else {
        k_full[idx] = k;
        v_full[idx] = v;
    }
}

// =====================================================================
// Host-side launcher
// =====================================================================
static inline dim3 gemm_grid(int M, int N, int batch)
{
    return dim3((N + BN - 1) / BN, (M + BM - 1) / BM, batch);
}

extern "C" void kernel_launch(void* const* d_in, const int* in_sizes, int n_in,
                              void* d_out, int out_size, void* d_ws, size_t ws_size,
                              hipStream_t stream)
{
    (void)in_sizes; (void)n_in; (void)out_size; (void)ws_size;
    const float* x        = (const float*)d_in[0];
    const float* id_total = (const float*)d_in[1];
    const float* mem_k    = (const float*)d_in[2];
    const float* mem_v    = (const float*)d_in[3];
    const float* w_qkv    = (const float*)d_in[4];
    const float* w_proj   = (const float*)d_in[5];
    const float* b_proj   = (const float*)d_in[6];
    const float* w_idkv   = (const float*)d_in[7];
    const float* b_idkv   = (const float*)d_in[8];

    float* out    = (float*)d_out;
    float* km_out = out + (long)B_ * N_ * C_;
    float* vm_out = km_out + (long)B_ * H_ * NM * Dh;

    float* ws = (float*)d_ws;
    size_t off = 0;
    float* qkv_ws  = ws + off; off += (size_t)B_ * N_ * 3 * C_;
    float* idkv_ws = ws + off; off += (size_t)B_ * NM * (C_ + H_);
    float* q_all   = ws + off; off += (size_t)B_ * H_ * N_ * Dh;
    float* k_full  = ws + off; off += (size_t)B_ * H_ * KF * Dh;
    float* v_full  = ws + off; off += (size_t)B_ * H_ * KF * Dh;
    float* attn_m  = ws + off; off += (size_t)B_ * H_ * NM * NM;
    float* attn_s  = ws + off; off += (size_t)B_ * H_ * NS * KF;
    float* xcat    = ws + off; off += (size_t)B_ * N_ * C_;
    float* tdm_snk = ws + off; off += 32;

    // 0) TDM warm of the memory-bank keys (tensor_load_to_lds + tensorcnt)
    tdm_stage_kernel<<<1, 32, 0, stream>>>(mem_k, tdm_snk);

    // 1) qkv = x @ w_qkv^T
    gemm_wmma_bf16<<<gemm_grid(B_ * N_, 3 * C_, 1), 256, 0, stream>>>(
        x, w_qkv, nullptr, qkv_ws,
        B_ * N_, 3 * C_, C_, C_, C_, 3 * C_,
        0, 0, 0, 0, 0, 0, 1, 1);

    // 2) id_kv = id_total @ w_idkv^T + b_idkv
    gemm_wmma_bf16<<<gemm_grid(B_ * NM, C_ + H_, 1), 256, 0, stream>>>(
        id_total, w_idkv, b_idkv, idkv_ws,
        B_ * NM, C_ + H_, C_, C_, C_, C_ + H_,
        0, 0, 0, 0, 0, 0, 1, 1);

    // 3/4) q split+scale; modulated k/v + full K/V banks
    {
        long tq = (long)B_ * H_ * N_ * Dh;
        prep_q_kernel<<<(unsigned)((tq + 255) / 256), 256, 0, stream>>>(qkv_ws, q_all);
        long tk = (long)B_ * H_ * KF * Dh;
        prep_kv_kernel<<<(unsigned)((tk + 255) / 256), 256, 0, stream>>>(
            qkv_ws, idkv_ws, mem_k, mem_v, k_full, v_full, km_out, vm_out);
    }

    // 5) attn_m = q_m @ k_m_mod^T (24 batches, 1682x1682, K=64)
    gemm_wmma_bf16<<<gemm_grid(NM, NM, B_ * H_), 256, 0, stream>>>(
        q_all, km_out, nullptr, attn_m,
        NM, NM, Dh, Dh, Dh, NM,
        (long)H_ * N_ * Dh, (long)N_ * Dh,
        (long)H_ * NM * Dh, (long)NM * Dh,
        (long)H_ * NM * NM, (long)NM * NM,
        H_, 1);

    // 6) top-k softmax (841 of 1682)
    topk_softmax_kernel<<<B_ * H_ * NM, 256, 0, stream>>>(attn_m, NM, TOPM);

    // 7) x_m = attn_m @ v_m_mod -> xcat[b, 0:NM, h*64:]
    gemm_wmma_bf16<<<gemm_grid(NM, Dh, B_ * H_), 256, 0, stream>>>(
        attn_m, vm_out, nullptr, xcat,
        NM, Dh, NM, NM, Dh, C_,
        (long)H_ * NM * NM, (long)NM * NM,
        (long)H_ * NM * Dh, (long)NM * Dh,
        (long)N_ * C_, (long)Dh,
        H_, 0);

    // 8) attn_s = q_s @ k_full^T (24 batches, 841x6619, K=64)
    gemm_wmma_bf16<<<gemm_grid(NS, KF, B_ * H_), 256, 0, stream>>>(
        q_all + (long)NM * Dh, k_full, nullptr, attn_s,
        NS, KF, Dh, Dh, Dh, KF,
        (long)H_ * N_ * Dh, (long)N_ * Dh,
        (long)H_ * KF * Dh, (long)KF * Dh,
        (long)H_ * NS * KF, (long)NS * KF,
        H_, 1);

    // 9) top-k softmax (1654 of 6619)
    topk_softmax_kernel<<<B_ * H_ * NS, 256, 0, stream>>>(attn_s, KF, TOPS);

    // 10) x_s = attn_s @ v_full -> xcat[b, NM:N, h*64:]
    gemm_wmma_bf16<<<gemm_grid(NS, Dh, B_ * H_), 256, 0, stream>>>(
        attn_s, v_full, nullptr, xcat + (long)NM * C_,
        NS, Dh, KF, KF, Dh, C_,
        (long)H_ * NS * KF, (long)NS * KF,
        (long)H_ * KF * Dh, (long)KF * Dh,
        (long)N_ * C_, (long)Dh,
        H_, 0);

    // 11) out = xcat @ w_proj^T + b_proj
    gemm_wmma_bf16<<<gemm_grid(B_ * N_, C_, 1), 256, 0, stream>>>(
        xcat, w_proj, b_proj, out,
        B_ * N_, C_, C_, C_, C_, C_,
        0, 0, 0, 0, 0, 0, 1, 1);
}
